// MultiHeadSelfAttention_53730040873227
// MI455X (gfx1250) — matrix operand order repaired and checked by reading, not compiled, not run on hardware
//
#include <hip/hip_runtime.h>
#include <hip/hip_bf16.h>

typedef _Float16 v16h __attribute__((ext_vector_type(16)));
typedef _Float16 v8h  __attribute__((ext_vector_type(8)));
typedef float    v8f  __attribute__((ext_vector_type(8)));

#define WMMA_F16(A, Bm, Cacc) \
    __builtin_amdgcn_wmma_f32_16x16x32_f16(false, (A), false, (Bm), (short)0, (Cacc), false, false)

static constexpr int Bc = 4, Cc = 2048, Dc = 1024, Hc = 16, HDc = 64;
static constexpr int QTILE = 16;              // query rows per block (attention kernel)
static constexpr int NKT   = Cc / 16;         // 128 key tiles
static constexpr float SCALE = 0.125f;        // 1/sqrt(64)
static constexpr float INV_H = 1.0f / 16.0f;

// ---------------------------------------------------------------------------
// Kernel 1: per-head QKV projection. Block = 256 thr handles (b, h, 128 rows).
// Weights are staged ONCE per block into LDS, converted f32->f16 and swizzled
// into WMMA B-fragment layout (lane = {g,col}, k = 8g+s, 16+8g+(s-8)), so the
// per-wave fragment fetch is two ds_load_b128 instead of 16 strided dwords.
// Q stored f16 [B,H,C,64] (pre-scaled by 1/sqrt(hd)), K f16 [B,H,C,64],
// V stored TRANSPOSED f16 [B,H,64,C] so that PV B-fragments are contiguous.
// ---------------------------------------------------------------------------
__global__ __launch_bounds__(256)
void qkv_proj_kernel(const float* __restrict__ x,
                     const float* __restrict__ Wq, const float* __restrict__ bq,
                     const float* __restrict__ Wk, const float* __restrict__ bk,
                     const float* __restrict__ Wv, const float* __restrict__ bv,
                     _Float16* __restrict__ Qb, _Float16* __restrict__ Kb,
                     _Float16* __restrict__ Vt) {
    // [mat][win][nt][lane][16] : 3*2*4*32*16 f16 = 24 KB
    __shared__ _Float16 sW[3 * 2 * 4 * 32 * 16];
    __shared__ float    sB[3 * 64];

    const int bz   = blockIdx.z;          // batch
    const int h    = blockIdx.y;          // head
    const int tid  = threadIdx.x;
    const int wave = tid >> 5;
    const int lane = tid & 31;
    const int g    = lane >> 4;           // half-wave group
    const int ln   = lane & 15;
    const int m0   = blockIdx.x * 128 + wave * 16;   // first row of this wave's tile

    const float* Wsrc[3] = { Wq + (size_t)h * 64 * 64, Wk + (size_t)h * 64 * 64,
                             Wv + (size_t)h * 64 * 64 };
    const float* bsrc[3] = { bq + h * 64, bk + h * 64, bv + h * 64 };

    // ---- cooperative, coalesced W stage: global f32 -> LDS f16 fragments ----
    // idx = (m, d, e) with e fastest -> coalesced global reads.
    for (int idx = tid; idx < 3 * 64 * 64; idx += 256) {
        const int e  = idx & 63;
        const int d  = (idx >> 6) & 63;
        const int m  = idx >> 12;
        const float v = Wsrc[m][d * 64 + e];
        // B-fragment position: win = d>>5, j = (d&7)|(((d>>4)&1)<<3), grp = (d>>3)&1,
        // nt = e>>4, col = e&15, lane = grp*16+col
        const int win = d >> 5, j = (d & 7) | (((d >> 4) & 1) << 3), grp = (d >> 3) & 1;
        const int nt  = e >> 4, col = e & 15;
        sW[((((m * 2 + win) * 4 + nt) * 32) + grp * 16 + col) * 16 + j] = (_Float16)v;
    }
    if (tid < 3 * 64) sB[tid] = bsrc[tid >> 6][tid & 63];
    __syncthreads();

    // ---- A fragments: rows of x for this head, converted f32 -> f16 ----
    // A layout (16-bit, 16x32): lane row = lane%16, k order [g*8..+8, 16+g*8..+8]
    const float* xr = x + ((size_t)bz * Cc + (m0 + ln)) * Dc + h * HDc;
    v16h a0, a1;
#pragma unroll
    for (int j = 0; j < 8; ++j) {
        a0[j]     = (_Float16)xr[g * 8 + j];
        a0[8 + j] = (_Float16)xr[16 + g * 8 + j];
        a1[j]     = (_Float16)xr[32 + g * 8 + j];
        a1[8 + j] = (_Float16)xr[48 + g * 8 + j];
    }

    const size_t hb = (size_t)(bz * Hc + h);

#pragma unroll
    for (int m = 0; m < 3; ++m) {
#pragma unroll
        for (int nt = 0; nt < 4; ++nt) {
            const int n = nt * 16 + ln;
            const v16h b0 = *(const v16h*)(sW + (((m * 2 + 0) * 4 + nt) * 32 + lane) * 16);
            const v16h b1 = *(const v16h*)(sW + (((m * 2 + 1) * 4 + nt) * 32 + lane) * 16);
            const float bias = sB[m * 64 + n];
            v8f acc = { bias, bias, bias, bias, bias, bias, bias, bias };
            acc = WMMA_F16(a0, b0, acc);
            acc = WMMA_F16(a1, b1, acc);
            // C/D layout: row = r + 8*g, col = n
            if (m == 2) {
                // V transposed: lane's 8 rows are CONTIGUOUS in Vt -> one b128 store
                v8h vv;
#pragma unroll
                for (int r = 0; r < 8; ++r) vv[r] = (_Float16)acc[r];
                *(v8h*)(Vt + (hb * 64 + n) * Cc + m0 + 8 * g) = vv;
            } else {
#pragma unroll
                for (int r = 0; r < 8; ++r) {
                    const int row = m0 + r + 8 * g;
                    if (m == 0) Qb[(hb * Cc + row) * 64 + n] = (_Float16)(acc[r] * SCALE);
                    else        Kb[(hb * Cc + row) * 64 + n] = (_Float16)acc[r];
                }
            }
        }
    }
}

// ---------------------------------------------------------------------------
// Kernel 2: attention. Block = 256 thr (8 waves) handles one (b, 16-row qtile)
// across all 16 heads. LDS: scores 16x2048 f32 (128KB) + attn-mean accumulator
// 16x2048 f32 (128KB) + 16x64 f32 reduce staging (4KB) = 260KB < 320KB/WGP.
// ---------------------------------------------------------------------------
__global__ __launch_bounds__(256)
void attn_kernel(const _Float16* __restrict__ Qb, const _Float16* __restrict__ Kb,
                 const _Float16* __restrict__ Vt,
                 float* __restrict__ out, float* __restrict__ amean) {
    extern __shared__ float smem[];
    float* sS   = smem;                       // [16][2048] scores -> probs
    float* sA   = smem + QTILE * Cc;          // [16][2048] attn-mean accumulator
    float* sRed = smem + 2 * QTILE * Cc;      // [16][64]   cross-wave reduce

    const int bz   = blockIdx.y;
    const int qt   = blockIdx.x;              // query tile
    const int tid  = threadIdx.x;
    const int wave = tid >> 5;
    const int lane = tid & 31;
    const int g    = lane >> 4;
    const int ln   = lane & 15;

    for (int i = tid; i < QTILE * Cc; i += 256) sA[i] = 0.0f;
    __syncthreads();

    for (int h = 0; h < Hc; ++h) {
        const size_t hb = (size_t)(bz * Hc + h);

        // ---- Phase 1: S = Q Kt  (scale pre-folded into Q) ----
        const _Float16* qp = Qb + (hb * Cc + qt * QTILE + ln) * 64;
        v16h a0, a1;
#pragma unroll
        for (int j = 0; j < 8; ++j) {
            a0[j]     = qp[g * 16 + j];
            a0[8 + j] = qp[g * 16 + 8 + j];
            a1[j]     = qp[32 + g * 16 + j];
            a1[8 + j] = qp[32 + g * 16 + 8 + j];
        }
        for (int kt = wave; kt < NKT; kt += 8) {
            const _Float16* kp = Kb + (hb * Cc + kt * 16 + ln) * 64;
            v16h b0 = *(const v16h*)(kp + g * 16);        // contiguous 32B/lane
            v16h b1 = *(const v16h*)(kp + 32 + g * 16);
            v8f acc = {};
            acc = WMMA_F16(a0, b0, acc);
            acc = WMMA_F16(a1, b1, acc);
            const int col = kt * 16 + ln;
#pragma unroll
            for (int r = 0; r < 8; ++r) sS[(r + 8 * g) * Cc + col] = acc[r];
        }
        __syncthreads();

        // ---- Phase 2: row softmax + attn-mean accumulation ----
        {
            const int row = tid >> 4;                     // 16 threads per row
            const int l4  = tid & 15;
            float* rp = sS + row * Cc;
            float mx = -3.4e38f;
            for (int j = l4; j < Cc; j += 16) mx = fmaxf(mx, rp[j]);
#pragma unroll
            for (int off = 1; off < 16; off <<= 1) mx = fmaxf(mx, __shfl_xor(mx, off, 16));
            float sum = 0.0f;
            for (int j = l4; j < Cc; j += 16) sum += __expf(rp[j] - mx);
#pragma unroll
            for (int off = 1; off < 16; off <<= 1) sum += __shfl_xor(sum, off, 16);
            const float inv = 1.0f / sum;
            float* ap = sA + row * Cc;
            for (int j = l4; j < Cc; j += 16) {
                const float p = __expf(rp[j] - mx) * inv;
                rp[j] = p;
                ap[j] += p * INV_H;
            }
        }
        __syncthreads();

        // ---- Phase 3: O = P V  (waves: 4 n-tiles x 2 k-halves) ----
        const int nt = wave & 3;
        const int kh = wave >> 2;
        const _Float16* vp = Vt + (hb * 64 + nt * 16 + ln) * Cc;  // row e of V^T
        v8f acc = {};
        for (int ck = kh * 32; ck < kh * 32 + 32; ++ck) {
            const int k0 = ck * 32;
            const float* srow = sS + ln * Cc + k0;
            v16h a;
#pragma unroll
            for (int j = 0; j < 8; ++j) {
                a[j]     = (_Float16)srow[g * 16 + j];
                a[8 + j] = (_Float16)srow[g * 16 + 8 + j];
            }
            v16h bb = *(const v16h*)(vp + k0 + g * 16);   // contiguous keys
            acc = WMMA_F16(a, bb, acc);
        }
        __syncthreads();
        if (wave >= 4) {
#pragma unroll
            for (int r = 0; r < 8; ++r)
                sRed[(r + 8 * g) * 64 + nt * 16 + ln] = acc[r];
        }
        __syncthreads();
        if (wave < 4) {
#pragma unroll
            for (int r = 0; r < 8; ++r) {
                const float v = acc[r] + sRed[(r + 8 * g) * 64 + nt * 16 + ln];
                const int row = qt * QTILE + r + 8 * g;
                out[((size_t)bz * Cc + row) * Dc + h * HDc + nt * 16 + ln] = v;
            }
        }
        __syncthreads();
    }

    // ---- attn_mean [B,C,C] ----
    for (int i = tid; i < QTILE * Cc; i += 256) {
        const int row = i >> 11, col = i & (Cc - 1);
        amean[((size_t)bz * Cc + qt * QTILE + row) * Cc + col] = sA[i];
    }
}

// ---------------------------------------------------------------------------
extern "C" void kernel_launch(void* const* d_in, const int* in_sizes, int n_in,
                              void* d_out, int out_size, void* d_ws, size_t ws_size,
                              hipStream_t stream) {
    const float* x  = (const float*)d_in[0];
    const float* Wq = (const float*)d_in[1];
    const float* bq = (const float*)d_in[2];
    const float* Wk = (const float*)d_in[3];
    const float* bk = (const float*)d_in[4];
    const float* Wv = (const float*)d_in[5];
    const float* bv = (const float*)d_in[6];

    const size_t per = (size_t)Bc * Hc * Cc * HDc;     // 8,388,608 f16 elements
    _Float16* Qb = (_Float16*)d_ws;
    _Float16* Kb = Qb + per;
    _Float16* Vt = Kb + per;                           // 48 MB total workspace

    float* out   = (float*)d_out;                      // [B,C,D]
    float* amean = out + (size_t)Bc * Cc * Dc;         // [B,C,C]

    dim3 gQ(Cc / 128, Hc, Bc);                         // 16 x 16 x 4 blocks
    qkv_proj_kernel<<<gQ, 256, 0, stream>>>(x, Wq, bq, Wk, bk, Wv, bv, Qb, Kb, Vt);

    const size_t shmem = (size_t)(2 * QTILE * Cc + QTILE * 64) * sizeof(float); // 260KB
    // idempotent & deterministic: set every call (no static guards allowed)
    hipFuncSetAttribute(reinterpret_cast<const void*>(attn_kernel),
                        hipFuncAttributeMaxDynamicSharedMemorySize, (int)shmem);
    dim3 gA(Cc / QTILE, Bc);                           // 128 x 4 blocks
    attn_kernel<<<gA, 256, shmem, stream>>>(Qb, Kb, Vt, out, amean);
}
